// LGAN_generator_10204842295652
// MI455X (gfx1250) — matrix operand order repaired and checked by reading, not compiled, not run on hardware
//
#include <hip/hip_runtime.h>
#include <hip/hip_bf16.h>

// ---------------------------------------------------------------------------
// CDNA5 (gfx1250) implementation of the CTGAN-style LSTM generator.
// GEMMs on v_wmma_f32_16x16x32_f16; LSTM weights staged through LDS with
// async global->LDS copies (double buffered, ASYNCcnt-tracked).
// ---------------------------------------------------------------------------

typedef _Float16 half_t;
typedef _Float16 v16h __attribute__((ext_vector_type(16)));
typedef float    v8f  __attribute__((ext_vector_type(8)));
typedef int      vsi4 __attribute__((vector_size(16)));   // matches builtin V4i

#define BATCH 8192
#define ZDIM  256
#define FDIM  512
#define LDIM  1024
#define KXDIM 768      // ZDIM + FDIM
#define GDIM  4096     // 4*LDIM
#define OUTW  104      // total output width

#define NSTEPS1 (KXDIM / 32)            // 24 K-steps on Wih
#define NSTEPS2 (LDIM / 32)             // 32 K-steps on Whh
#define NSTEPS  (NSTEPS1 + NSTEPS2)     // 56
#define BROWS   256                     // weight rows staged per K-step
#define BSTRIDE 40                      // halfs per LDS row (64B data + 16B pad)

#if defined(__has_builtin)
#if __has_builtin(__builtin_amdgcn_global_load_async_to_lds_b128)
#define HAVE_ASYNC_LDS 1
#endif
#if __has_builtin(__builtin_amdgcn_s_wait_asynccnt)
#define HAVE_WAIT_ASYNC 1
#endif
#endif

#define AS1 __attribute__((address_space(1)))
#define AS3 __attribute__((address_space(3)))

__device__ __forceinline__ void wait_asynccnt0() {
#if defined(HAVE_WAIT_ASYNC)
  __builtin_amdgcn_s_wait_asynccnt(0);
#else
  asm volatile("s_wait_asynccnt 0x0" ::: "memory");
#endif
}

__device__ __forceinline__ v8f wmma16(v16h a, v16h b, v8f c) {
  return __builtin_amdgcn_wmma_f32_16x16x32_f16(false, a, false, b,
                                                (short)0, c, false, false);
}

// A-matrix fragment (16x32, MxK), row-major A with leading dim lda.
__device__ __forceinline__ v16h load_a_frag(const half_t* __restrict__ A, int lda,
                                            int m0, int k0, int lane) {
  int hs = lane >> 4;
  int m  = m0 + (lane & 15);
  const half_t* p = A + (size_t)m * lda + k0 + hs * 16;
  v16h r;
  *reinterpret_cast<float4*>(&r)       = *reinterpret_cast<const float4*>(p);
  *(reinterpret_cast<float4*>(&r) + 1) = *reinterpret_cast<const float4*>(p + 8);
  return r;
}

// B-matrix fragment (32x16) from global W [N,K] row-major.
__device__ __forceinline__ v16h load_b_frag(const half_t* __restrict__ W, int ldw,
                                            int n0, int k0, int lane) {
  int hs = lane >> 4;
  int n  = n0 + (lane & 15);
  const half_t* p = W + (size_t)n * ldw + k0 + hs * 16;
  v16h r;
  *reinterpret_cast<float4*>(&r)       = *reinterpret_cast<const float4*>(p);
  *(reinterpret_cast<float4*>(&r) + 1) = *reinterpret_cast<const float4*>(p + 8);
  return r;
}

// B-matrix fragment from the LDS-staged tile (row = g*64 + local j).
__device__ __forceinline__ v16h load_b_lds(const half_t* base, int jl, int g, int lane) {
  int row = g * 64 + jl + (lane & 15);
  const half_t* p = base + row * BSTRIDE + (lane >> 4) * 16;
  v16h r;
  *reinterpret_cast<float4*>(&r)       = *reinterpret_cast<const float4*>(p);
  *(reinterpret_cast<float4*>(&r) + 1) = *reinterpret_cast<const float4*>(p + 8);
  return r;
}

__device__ __forceinline__ float sigm_f(float x) {
  return 1.0f / (1.0f + __expf(-x));
}
__device__ __forceinline__ float tanh_f(float x) {
  x = fminf(fmaxf(x, -15.0f), 15.0f);
  float e = __expf(2.0f * x);
  return (e - 1.0f) / (e + 1.0f);
}

// ---------------------------------------------------------------------------
// Weight packing: f32 [N,K] -> f16 [Npad,Kpad], zero filled.
// ---------------------------------------------------------------------------
__global__ void pack_weight_kernel(const float* __restrict__ src, half_t* __restrict__ dst,
                                   int N, int K, int Npad, int Kpad) {
  int idx = blockIdx.x * blockDim.x + threadIdx.x;
  int total = Npad * Kpad;
  if (idx >= total) return;
  int n = idx / Kpad, k = idx - n * Kpad;
  float v = (n < N && k < K) ? src[(size_t)n * K + k] : 0.0f;
  dst[idx] = (half_t)v;
}

__global__ void bias_sum_kernel(const float* __restrict__ bih, const float* __restrict__ bhh,
                                float* __restrict__ bsum) {
  int i = blockIdx.x * blockDim.x + threadIdx.x;
  if (i < GDIM) bsum[i] = bih[i] + bhh[i];
}

__global__ void init_xcat_kernel(const float* __restrict__ z, const float* __restrict__ go,
                                 half_t* __restrict__ xcat) {
  int idx = blockIdx.x * blockDim.x + threadIdx.x;
  if (idx >= BATCH * KXDIM) return;
  int row = idx / KXDIM, col = idx - row * KXDIM;
  float v = (col < ZDIM) ? z[(size_t)row * ZDIM + col] : go[col - ZDIM];
  xcat[idx] = (half_t)v;
}

__global__ void init_state_kernel(const float* __restrict__ hx0, const float* __restrict__ cx0,
                                  half_t* __restrict__ h16, float* __restrict__ c32) {
  int idx = blockIdx.x * blockDim.x + threadIdx.x;
  if (idx >= BATCH * LDIM) return;
  h16[idx] = (half_t)hx0[idx];
  c32[idx] = cx0[idx];
}

// ---------------------------------------------------------------------------
// Fused LSTM cell with LDS-staged weights.
//   gates = xcat @ Wih^T + h_in @ Whh^T + (bih + bhh)
//   c = sigm(f)*c + sigm(i)*tanh(g);  h_out = sigm(o)*tanh(c)
// Block: 256 thr = 8 waves, block tile 64(M) x 64(j); wave 16x32, 4 gates.
// Each K-step stages 256 weight rows x 64B into LDS (async, double buffered).
// ---------------------------------------------------------------------------
__global__ __launch_bounds__(256)
void lstm_kernel(const half_t* __restrict__ xcat, const half_t* __restrict__ hin,
                 float* __restrict__ c32, half_t* __restrict__ hout,
                 const half_t* __restrict__ wih, const half_t* __restrict__ whh,
                 const float* __restrict__ bsum) {
  __shared__ half_t ldsB[2][BROWS * BSTRIDE];

  int tid = threadIdx.x, wave = tid >> 5, lane = tid & 31;
  int wm = wave >> 1, wn = wave & 1;
  int m0 = blockIdx.x * 64 + wm * 16;
  int jb0 = blockIdx.y * 64;
  int jloc = wn * 32;

  // staging assignment: thread t copies W row (g*1024 + jb0 + j), g=t>>6, j=t&63
  int sg = tid >> 6, sj = tid & 63;

  v8f acc[4][2] = {};

  auto src_for_step = [&](int s) -> const half_t* {
    if (s < NSTEPS1)
      return wih + (size_t)(sg * LDIM + jb0 + sj) * KXDIM + s * 32;
    return whh + (size_t)(sg * LDIM + jb0 + sj) * LDIM + (s - NSTEPS1) * 32;
  };

#if defined(HAVE_ASYNC_LDS)
  auto stage = [&](int s, int buf) {
    const half_t* src = src_for_step(s);
    half_t* dst = &ldsB[buf][tid * BSTRIDE];
#pragma unroll
    for (int c = 0; c < 4; ++c) {
      __builtin_amdgcn_global_load_async_to_lds_b128(
          (AS1 vsi4*)(src + c * 8), (AS3 vsi4*)(dst + c * 8), 0, 0);
    }
  };

  int p = 0;
  // Per-step body: wait for buffer p, prefetch next tile into p^1, then
  // preload all 8 B fragments (independent registers) and run 8 WMMAs.
  auto body = [&](const half_t* A, int lda, int k0, int s) {
    wait_asynccnt0();      // my staging ops for buffer p are done
    __syncthreads();       // everyone's staging done; prior reads of p^1 done
    if (s + 1 < NSTEPS) stage(s + 1, p ^ 1);

    v16h a = load_a_frag(A, lda, m0, k0, lane);
    const half_t* bb = &ldsB[p][0];
    v16h bfr[2][4];
#pragma unroll
    for (int nt = 0; nt < 2; ++nt)
#pragma unroll
      for (int g = 0; g < 4; ++g)
        bfr[nt][g] = load_b_lds(bb, jloc + nt * 16, g, lane);
#pragma unroll
    for (int nt = 0; nt < 2; ++nt)
#pragma unroll
      for (int g = 0; g < 4; ++g)
        acc[g][nt] = wmma16(a, bfr[nt][g], acc[g][nt]);
    p ^= 1;
  };

  stage(0, 0);
  for (int ks = 0; ks < NSTEPS1; ++ks)
    body(xcat, KXDIM, ks * 32, ks);
  for (int ks = 0; ks < NSTEPS2; ++ks)
    body(hin, LDIM, ks * 32, NSTEPS1 + ks);
#else
  // Fallback: global->VGPR->LDS staging, single buffer, two barriers per step.
  float4 rr[4];
  {
    const half_t* src = src_for_step(0);
#pragma unroll
    for (int c = 0; c < 4; ++c)
      rr[c] = *reinterpret_cast<const float4*>(src + c * 8);
  }
  for (int s = 0; s < NSTEPS; ++s) {
    __syncthreads();  // previous step's LDS reads complete
    half_t* dst = &ldsB[0][tid * BSTRIDE];
#pragma unroll
    for (int c = 0; c < 4; ++c)
      *reinterpret_cast<float4*>(dst + c * 8) = rr[c];
    if (s + 1 < NSTEPS) {
      const half_t* src = src_for_step(s + 1);
#pragma unroll
      for (int c = 0; c < 4; ++c)
        rr[c] = *reinterpret_cast<const float4*>(src + c * 8);
    }
    __syncthreads();  // staged tile visible

    const half_t* A;
    int lda, k0;
    if (s < NSTEPS1) { A = xcat; lda = KXDIM; k0 = s * 32; }
    else             { A = hin;  lda = LDIM;  k0 = (s - NSTEPS1) * 32; }
    v16h a = load_a_frag(A, lda, m0, k0, lane);

    const half_t* bb = &ldsB[0][0];
    v16h bfr[2][4];
#pragma unroll
    for (int nt = 0; nt < 2; ++nt)
#pragma unroll
      for (int g = 0; g < 4; ++g)
        bfr[nt][g] = load_b_lds(bb, jloc + nt * 16, g, lane);
#pragma unroll
    for (int nt = 0; nt < 2; ++nt)
#pragma unroll
      for (int g = 0; g < 4; ++g)
        acc[g][nt] = wmma16(a, bfr[nt][g], acc[g][nt]);
  }
#endif

  int l15 = lane & 15, hs = lane >> 4;
#pragma unroll
  for (int nt = 0; nt < 2; ++nt) {
    int j = jb0 + jloc + nt * 16 + l15;
    float bi = bsum[j], bf = bsum[LDIM + j], bg = bsum[2 * LDIM + j], bo = bsum[3 * LDIM + j];
#pragma unroll
    for (int r = 0; r < 8; ++r) {
      int m = m0 + r + hs * 8;
      float gi = acc[0][nt][r] + bi;
      float gf = acc[1][nt][r] + bf;
      float gg = acc[2][nt][r] + bg;
      float go = acc[3][nt][r] + bo;
      float i_ = sigm_f(gi), f_ = sigm_f(gf), g_ = tanh_f(gg), o_ = sigm_f(go);
      size_t ci = (size_t)m * LDIM + j;
      float c2 = f_ * c32[ci] + i_ * g_;
      c32[ci] = c2;
      hout[ci] = (half_t)(o_ * tanh_f(c2));
    }
  }
}

// ---------------------------------------------------------------------------
// Generic GEMM, f16 output: D = act(A[M,K] @ W[N,K]^T + bias), N=512.
// Block tile 64(M) x 128(N), 8 waves of 32x32. ACT: 0=none, 1=tanh.
// ---------------------------------------------------------------------------
template <int ACT>
__global__ __launch_bounds__(256)
void gemm_f16out_kernel(const half_t* __restrict__ A, int lda,
                        const half_t* __restrict__ W, int ldw,
                        const float* __restrict__ bias, int K,
                        half_t* __restrict__ D, int ldd) {
  int tid = threadIdx.x, wave = tid >> 5, lane = tid & 31;
  int wm = wave >> 2, wn = wave & 3;
  int m0 = blockIdx.x * 64 + wm * 32;
  int n0 = blockIdx.y * 128 + wn * 32;

  v8f acc[2][2] = {};
  for (int k0 = 0; k0 < K; k0 += 32) {
    v16h a0 = load_a_frag(A, lda, m0, k0, lane);
    v16h a1 = load_a_frag(A, lda, m0 + 16, k0, lane);
    v16h b0 = load_b_frag(W, ldw, n0, k0, lane);
    v16h b1 = load_b_frag(W, ldw, n0 + 16, k0, lane);
    acc[0][0] = wmma16(a0, b0, acc[0][0]);
    acc[1][0] = wmma16(a1, b0, acc[1][0]);
    acc[0][1] = wmma16(a0, b1, acc[0][1]);
    acc[1][1] = wmma16(a1, b1, acc[1][1]);
  }

  int l15 = lane & 15, hs = lane >> 4;
#pragma unroll
  for (int mt = 0; mt < 2; ++mt) {
#pragma unroll
    for (int nt = 0; nt < 2; ++nt) {
      int n = n0 + nt * 16 + l15;
      float bv = bias[n];
#pragma unroll
      for (int r = 0; r < 8; ++r) {
        int m = m0 + mt * 16 + r + hs * 8;
        float v = acc[mt][nt][r] + bv;
        if (ACT == 1) v = tanh_f(v);
        D[(size_t)m * ldd + n] = (half_t)v;
      }
    }
  }
}

// ---------------------------------------------------------------------------
// Small-N GEMM (projection heads): vpre[B,32] = A[M,512] @ Wpad[32,512]^T + bias.
// ---------------------------------------------------------------------------
__global__ __launch_bounds__(256)
void gemm_smalln_kernel(const half_t* __restrict__ A, int lda,
                        const half_t* __restrict__ W,     // [32, 512]
                        const float* __restrict__ bias, int Nreal,
                        float* __restrict__ vpre) {
  int tid = threadIdx.x, wave = tid >> 5, lane = tid & 31;
  int m0 = blockIdx.x * 256 + wave * 32;

  v8f acc[2][2] = {};
  for (int k0 = 0; k0 < FDIM; k0 += 32) {
    v16h a0 = load_a_frag(A, lda, m0, k0, lane);
    v16h a1 = load_a_frag(A, lda, m0 + 16, k0, lane);
    v16h b0 = load_b_frag(W, FDIM, 0, k0, lane);
    v16h b1 = load_b_frag(W, FDIM, 16, k0, lane);
    acc[0][0] = wmma16(a0, b0, acc[0][0]);
    acc[1][0] = wmma16(a1, b0, acc[1][0]);
    acc[0][1] = wmma16(a0, b1, acc[0][1]);
    acc[1][1] = wmma16(a1, b1, acc[1][1]);
  }

  int l15 = lane & 15, hs = lane >> 4;
#pragma unroll
  for (int mt = 0; mt < 2; ++mt) {
#pragma unroll
    for (int nt = 0; nt < 2; ++nt) {
      int n = nt * 16 + l15;
      float bv = (n < Nreal) ? bias[n] : 0.0f;
#pragma unroll
      for (int r = 0; r < 8; ++r) {
        int m = m0 + mt * 16 + r + hs * 8;
        vpre[(size_t)m * 32 + n] = acc[mt][nt][r] + bv;
      }
    }
  }
}

// ---------------------------------------------------------------------------
// Per-row activation + output scatter + f16 (padded) copy for next GEMM.
// act: 0=tanh, 1=sigmoid, 2=softmax
// ---------------------------------------------------------------------------
__global__ void act_kernel(const float* __restrict__ vpre, float* __restrict__ out,
                           int off, half_t* __restrict__ v16, int d, int act) {
  int row = blockIdx.x * blockDim.x + threadIdx.x;
  if (row >= BATCH) return;
  float vals[32];
  for (int i = 0; i < d; ++i) vals[i] = vpre[(size_t)row * 32 + i];
  if (act == 0) {
    for (int i = 0; i < d; ++i) vals[i] = tanh_f(vals[i]);
  } else if (act == 1) {
    for (int i = 0; i < d; ++i) vals[i] = sigm_f(vals[i]);
  } else {
    float mx = vals[0];
    for (int i = 1; i < d; ++i) mx = fmaxf(mx, vals[i]);
    float s = 0.0f;
    for (int i = 0; i < d; ++i) { vals[i] = __expf(vals[i] - mx); s += vals[i]; }
    float inv = 1.0f / s;
    for (int i = 0; i < d; ++i) vals[i] *= inv;
  }
  for (int i = 0; i < d; ++i) {
    out[(size_t)row * OUTW + off + i] = vals[i];
    v16[(size_t)row * 32 + i] = (half_t)vals[i];
  }
  for (int i = d; i < 32; ++i) v16[(size_t)row * 32 + i] = (half_t)0.0f;
}

// ---------------------------------------------------------------------------
// Host orchestration
// ---------------------------------------------------------------------------
namespace {

struct ColIdx {
  int w0, b0;        // fe0w/few, fe0b/feb
  int w1, b1;        // fe1w, fe1b (gmm only)
  int fc1w, fc1b, fc2w, fc2b, fc3w, fc3b;  // fc3 gmm only
};

struct InputMap {
  int z, hx0, cx0, Wih, Whh, bih, bhh, go;
  ColIdx col[8];
};

static const int  kColDim[8]  = {11, 32, 1, 11, 1, 5, 32, 11};
// 0=gmm, 1=onehot, 2=normalize, 3=binary, 4=ordinal
static const int  kColType[8] = {0, 1, 2, 0, 3, 4, 1, 0};
static const int  kOutOff[8]  = {0, 11, 43, 44, 55, 56, 61, 93};

static InputMap build_map(bool insertion_order) {
  InputMap m{};
  if (insertion_order) {
    m.z = 0; m.hx0 = 1; m.cx0 = 2;
    m.Wih = 3; m.Whh = 4; m.bih = 5; m.bhh = 6; m.go = 7;
    int i = 8;
    for (int c = 0; c < 8; ++c) {
      ColIdx& ci = m.col[c];
      if (kColType[c] == 0) {
        ci.w0 = i; ci.b0 = i + 1; ci.w1 = i + 2; ci.b1 = i + 3;
        ci.fc1w = i + 4; ci.fc1b = i + 5; ci.fc2w = i + 6; ci.fc2b = i + 7;
        ci.fc3w = i + 8; ci.fc3b = i + 9; i += 10;
      } else {
        ci.w0 = i; ci.b0 = i + 1;
        ci.fc1w = i + 2; ci.fc1b = i + 3; ci.fc2w = i + 4; ci.fc2b = i + 5;
        ci.w1 = -1; ci.b1 = -1; ci.fc3w = -1; ci.fc3b = -1; i += 6;
      }
    }
  } else {
    // jax tree_leaves: sorted dict keys at every level.
    m.cx0 = 0; m.hx0 = 1;
    int i = 2;
    for (int c = 0; c < 8; ++c) {
      ColIdx& ci = m.col[c];
      if (kColType[c] == 0) {
        ci.fc1b = i; ci.fc1w = i + 1; ci.fc2b = i + 2; ci.fc2w = i + 3;
        ci.fc3b = i + 4; ci.fc3w = i + 5;
        ci.b0 = i + 6; ci.w0 = i + 7; ci.b1 = i + 8; ci.w1 = i + 9; i += 10;
      } else {
        ci.fc1b = i; ci.fc1w = i + 1; ci.fc2b = i + 2; ci.fc2w = i + 3;
        ci.b0 = i + 4; ci.w0 = i + 5;
        ci.w1 = -1; ci.b1 = -1; ci.fc3w = -1; ci.fc3b = -1; i += 6;
      }
    }
    m.go = i;
    m.Whh = i + 1; m.Wih = i + 2;
    m.bhh = i + 3; m.bih = i + 4;
    m.z = i + 5;
  }
  return m;
}

}  // namespace

extern "C" void kernel_launch(void* const* d_in, const int* in_sizes, int n_in,
                              void* d_out, int out_size, void* d_ws, size_t ws_size,
                              hipStream_t stream) {
  (void)n_in; (void)out_size; (void)ws_size;
  const bool insertion = (in_sizes[0] == BATCH * ZDIM);
  InputMap map = build_map(insertion);
  auto in = [&](int idx) { return (const float*)d_in[idx]; };

  // ---- workspace carve-out -------------------------------------------------
  char* ws = (char*)d_ws;
  size_t off = 0;
  auto carve = [&](size_t bytes) {
    size_t r = off;
    off = (off + bytes + 255) & ~(size_t)255;
    return r;
  };
  half_t* wih16 = (half_t*)(ws + carve((size_t)GDIM * KXDIM * 2));
  half_t* whh16 = (half_t*)(ws + carve((size_t)GDIM * LDIM * 2));
  float*  bsum  = (float*) (ws + carve((size_t)GDIM * 4));
  half_t* xcat  = (half_t*)(ws + carve((size_t)BATCH * KXDIM * 2));
  half_t* hA    = (half_t*)(ws + carve((size_t)BATCH * LDIM * 2));
  half_t* hB    = (half_t*)(ws + carve((size_t)BATCH * LDIM * 2));
  float*  c32   = (float*) (ws + carve((size_t)BATCH * LDIM * 4));
  half_t* v16   = (half_t*)(ws + carve((size_t)BATCH * 32 * 2));
  float*  vpre  = (float*) (ws + carve((size_t)BATCH * 32 * 4));
  half_t* feA[8]; half_t* feB[8]; half_t* fc1p[8]; half_t* fc2p[8]; half_t* fc3p[8];
  for (int c = 0; c < 8; ++c) {
    feA[c]  = (half_t*)(ws + carve((size_t)FDIM * LDIM * 2));
    feB[c]  = (half_t*)(ws + carve((size_t)FDIM * LDIM * 2));
    fc1p[c] = (half_t*)(ws + carve((size_t)32 * FDIM * 2));
    fc2p[c] = (half_t*)(ws + carve((size_t)32 * FDIM * 2));
    fc3p[c] = (half_t*)(ws + carve((size_t)FDIM * 32 * 2));
  }

  auto pack = [&](const float* src, half_t* dst, int N, int K, int Npad, int Kpad) {
    int total = Npad * Kpad;
    pack_weight_kernel<<<(total + 255) / 256, 256, 0, stream>>>(src, dst, N, K, Npad, Kpad);
  };

  // ---- weight conversion (every call; deterministic) -----------------------
  pack(in(map.Wih), wih16, GDIM, KXDIM, GDIM, KXDIM);
  pack(in(map.Whh), whh16, GDIM, LDIM, GDIM, LDIM);
  bias_sum_kernel<<<GDIM / 256, 256, 0, stream>>>(in(map.bih), in(map.bhh), bsum);

  for (int c = 0; c < 8; ++c) {
    const ColIdx& ci = map.col[c];
    int d = kColDim[c];
    if (kColType[c] == 0) {  // gmm
      pack(in(ci.w0),   feA[c],  FDIM, LDIM, FDIM, LDIM);
      pack(in(ci.w1),   feB[c],  FDIM, LDIM, FDIM, LDIM);
      pack(in(ci.fc1w), fc1p[c], 1,     FDIM, 32, FDIM);
      pack(in(ci.fc2w), fc2p[c], d - 1, FDIM, 32, FDIM);
      pack(in(ci.fc3w), fc3p[c], FDIM,  d - 1, FDIM, 32);
    } else {
      pack(in(ci.w0),   feA[c],  FDIM, LDIM, FDIM, LDIM);
      pack(in(ci.fc1w), fc1p[c], d,    FDIM, 32, FDIM);
      pack(in(ci.fc2w), fc3p[c], FDIM, d,    FDIM, 32);  // fc2 is [512,d] -> K-padded
    }
  }

  // ---- state init ----------------------------------------------------------
  init_xcat_kernel<<<(BATCH * KXDIM + 255) / 256, 256, 0, stream>>>(
      in(map.z), in(map.go), xcat);
  init_state_kernel<<<(BATCH * LDIM + 255) / 256, 256, 0, stream>>>(
      in(map.hx0), in(map.cx0), hA, c32);

  // ---- pipeline ------------------------------------------------------------
  float* out = (float*)d_out;
  half_t* fx = xcat + ZDIM;  // fx region, stride KXDIM
  half_t* hcur = hA;
  half_t* hnxt = hB;

  dim3 lstm_grid(BATCH / 64, LDIM / 64);
  dim3 fe_grid(BATCH / 64, FDIM / 128);

  auto do_lstm = [&]() {
    lstm_kernel<<<lstm_grid, 256, 0, stream>>>(xcat, hcur, c32, hnxt, wih16, whh16, bsum);
    half_t* t = hcur; hcur = hnxt; hnxt = t;
  };

  for (int c = 0; c < 8; ++c) {
    const ColIdx& ci = map.col[c];
    int d = kColDim[c];
    if (kColType[c] == 0) {  // ---- gmm ----
      do_lstm();
      gemm_f16out_kernel<1><<<fe_grid, 256, 0, stream>>>(
          hcur, LDIM, feA[c], LDIM, in(ci.b0), LDIM, fx, KXDIM);
      gemm_smalln_kernel<<<BATCH / 256, 256, 0, stream>>>(
          fx, KXDIM, fc1p[c], in(ci.fc1b), 1, vpre);
      act_kernel<<<BATCH / 256, 256, 0, stream>>>(vpre, out, kOutOff[c], v16, 1, 0);

      do_lstm();
      gemm_f16out_kernel<1><<<fe_grid, 256, 0, stream>>>(
          hcur, LDIM, feB[c], LDIM, in(ci.b1), LDIM, fx, KXDIM);
      gemm_smalln_kernel<<<BATCH / 256, 256, 0, stream>>>(
          fx, KXDIM, fc2p[c], in(ci.fc2b), d - 1, vpre);
      act_kernel<<<BATCH / 256, 256, 0, stream>>>(vpre, out, kOutOff[c] + 1, v16, d - 1, 2);

      gemm_f16out_kernel<1><<<fe_grid, 256, 0, stream>>>(
          v16, 32, fc3p[c], 32, in(ci.fc3b), 32, fx, KXDIM);
    } else {                 // ---- one-hot / normalize / binary / ordinal ----
      do_lstm();
      gemm_f16out_kernel<0><<<fe_grid, 256, 0, stream>>>(
          hcur, LDIM, feA[c], LDIM, in(ci.b0), LDIM, fx, KXDIM);
      gemm_smalln_kernel<<<BATCH / 256, 256, 0, stream>>>(
          fx, KXDIM, fc1p[c], in(ci.fc1b), d, vpre);
      int act = (kColType[c] == 1) ? 2 : (kColType[c] == 2) ? 0 : 1;
      act_kernel<<<BATCH / 256, 256, 0, stream>>>(vpre, out, kOutOff[c], v16, d, act);
      gemm_f16out_kernel<0><<<fe_grid, 256, 0, stream>>>(
          v16, 32, fc3p[c], 32, in(ci.fc2b), 32, fx, KXDIM);
    }
  }
}
